// ReflexMoELayer_83124797046960
// MI455X (gfx1250) — compile-verified, matrix operand order checked by reading, not run on hardware
//
#include <hip/hip_runtime.h>
#include <hip/hip_bf16.h>

// ---------------- problem constants ----------------
#define N_EXPERTS 8
#define TOP_K     2
#define D_MODEL   1024
#define D_FF      4096
#define T_TOKENS  4096          // 2 * 2048
#define TILE_M    16            // token tile
#define FCHUNK    256           // F columns per pipeline iteration

typedef float        v8f   __attribute__((ext_vector_type(8)));
typedef __bf16       v16bf __attribute__((ext_vector_type(16)));
typedef unsigned int u32;
typedef u32          v4u   __attribute__((ext_vector_type(4)));
typedef u32          v8u   __attribute__((ext_vector_type(8)));

#define WMMA_BF16(a, b, c) \
  __builtin_amdgcn_wmma_f32_16x16x32_bf16(false, (a), false, (b), (short)0, (c), false, false)

// ---------------- fragment loaders ----------------
// A fragment: 16x32 bf16 (MxK) from LDS. lanes 0-15: M=lane, K=kk..kk+15;
// lanes 16-31: same M, K=kk+16..kk+31 (contiguous k order, same as the B loader).
__device__ __forceinline__ v16bf load_a_lds(const __bf16* base, int ldk, int kk, int lane) {
  const int m = lane & 15, half = lane >> 4;
  const __bf16* p = base + m * ldk + kk + half * 16;
  v16bf a;
#pragma unroll
  for (int j = 0; j < 16; ++j) a[j] = p[j];
  return a;
}

// B fragment: 32x16 bf16 (KxN) from global fp32 weights, converting in-register.
__device__ __forceinline__ v16bf load_b_gf32(const float* base, long ldk, int kk, int lane) {
  const int n = lane & 15, half = lane >> 4;
  const float* p = base + (long)n * ldk + kk + half * 16;
  v16bf b;
#pragma unroll
  for (int j = 0; j < 16; j += 4) {
    float4 f = *(const float4*)(p + j);
    b[j + 0] = (__bf16)f.x; b[j + 1] = (__bf16)f.y;
    b[j + 2] = (__bf16)f.z; b[j + 3] = (__bf16)f.w;
  }
  return b;
}

__device__ __forceinline__ float fast_silu(float z) {
  // z * sigmoid(z) with v_rcp_f32 instead of IEEE divide expansion
  return z * __builtin_amdgcn_rcpf(1.f + __expf(-z));
}

// ---------------- kernel 0: zero out + counters ----------------
__global__ void moe_zero(float* __restrict__ out, int* __restrict__ cnt,
                         float* __restrict__ imp, int n4) {
  int i = blockIdx.x * blockDim.x + threadIdx.x;
  if (i < n4) ((float4*)out)[i] = make_float4(0.f, 0.f, 0.f, 0.f);
  if (i < N_EXPERTS) { cnt[i] = 0; imp[i] = 0.f; }
}

// ---------------- kernel 0b: x -> bf16 staging copy (TDM source) ----------------
__global__ void moe_cvt_bf16(const float* __restrict__ x, __bf16* __restrict__ xb, int n4) {
  int i = blockIdx.x * blockDim.x + threadIdx.x;
  if (i >= n4) return;
  float4 v = ((const float4*)x)[i];
  union { __bf16 h[4]; unsigned long long u; } p;
  p.h[0] = (__bf16)v.x; p.h[1] = (__bf16)v.y; p.h[2] = (__bf16)v.z; p.h[3] = (__bf16)v.w;
  ((unsigned long long*)xb)[i] = p.u;
}

// ---------------- kernel 1: router ----------------
__global__ void moe_router(const float* __restrict__ x, const float* __restrict__ Wr,
                           int* __restrict__ cnt, int* __restrict__ idx,
                           float* __restrict__ wgt, float* __restrict__ imp) {
  const int wave = threadIdx.x >> 5;
  const int lane = threadIdx.x & 31;
  const int t = blockIdx.x * 2 + wave;
  if (t >= T_TOKENS) return;

  const int e = lane & 7;
  const int chunk = lane >> 3;

  const float4* x4  = (const float4*)(x + (long)t * D_MODEL) + chunk * 64;
  const float4* wr4 = (const float4*)(Wr + (long)e * D_MODEL) + chunk * 64;
  float acc = 0.f;
#pragma unroll 4
  for (int i = 0; i < 64; ++i) {
    float4 a = x4[i], b = wr4[i];
    acc += a.x * b.x + a.y * b.y + a.z * b.z + a.w * b.w;
  }
  acc += __shfl_xor(acc, 8, 32);
  acc += __shfl_xor(acc, 16, 32);

  float lg[N_EXPERTS];
#pragma unroll
  for (int i = 0; i < N_EXPERTS; ++i) lg[i] = __shfl(acc, i, 32);

  float mx = lg[0];
#pragma unroll
  for (int i = 1; i < N_EXPERTS; ++i) mx = fmaxf(mx, lg[i]);
  float p[N_EXPERTS], s = 0.f;
#pragma unroll
  for (int i = 0; i < N_EXPERTS; ++i) { p[i] = __expf(lg[i] - mx); s += p[i]; }
  float inv = __builtin_amdgcn_rcpf(s);
#pragma unroll
  for (int i = 0; i < N_EXPERTS; ++i) p[i] *= inv;

  if (lane < N_EXPERTS) atomicAdd(&imp[lane], p[lane]);

  int i0 = 0;
#pragma unroll
  for (int i = 1; i < N_EXPERTS; ++i) if (p[i] > p[i0]) i0 = i;
  int i1 = (i0 == 0) ? 1 : 0;
#pragma unroll
  for (int i = 0; i < N_EXPERTS; ++i) if (i != i0 && p[i] > p[i1]) i1 = i;

  if (lane == 0) {
    float rs = 1.f / (p[i0] + p[i1]);
    int pos0 = atomicAdd(&cnt[i0], 1);
    idx[i0 * T_TOKENS + pos0] = t;
    wgt[i0 * T_TOKENS + pos0] = p[i0] * rs;
    int pos1 = atomicAdd(&cnt[i1], 1);
    idx[i1 * T_TOKENS + pos1] = t;
    wgt[i1 * T_TOKENS + pos1] = p[i1] * rs;
  }
}

// ---------------- kernel 2: fused expert FFN (TDM gather + WMMA bf16) ----------------
// grid.x = expert * (T/16) + tile ; block = 256 threads (8 waves)
__global__ void __launch_bounds__(256)
moe_expert_ffn(const __bf16* __restrict__ xb,
               const float* __restrict__ W1, const float* __restrict__ W2,
               const float* __restrict__ W3,
               const int* __restrict__ cnt, const int* __restrict__ idx,
               const float* __restrict__ wgt, float* __restrict__ out) {
  const int tilesPerExpert = T_TOKENS / TILE_M;     // 256
  const int e = blockIdx.x / tilesPerExpert;
  const int m = blockIdx.x % tilesPerExpert;
  const int cnt_e = cnt[e];
  if (m * TILE_M >= cnt_e) return;                   // uniform early-out

  __shared__ __bf16 xs[TILE_M][D_MODEL];             // 32 KB  x-tile (bf16, TDM dest)
  __shared__ __bf16 hs[TILE_M][FCHUNK];              // 8 KB   h-chunk (bf16)
  __shared__ int    tok_s[TILE_M];
  __shared__ float  wt_s[TILE_M];

  const int tid  = threadIdx.x;
  const int w    = tid >> 5;                         // wave id 0..7
  const int lane = tid & 31;

  if (tid < TILE_M) {
    int g = m * TILE_M + tid;
    int j = min(g, cnt_e - 1);
    tok_s[tid] = idx[e * T_TOKENS + j];
    wt_s[tid]  = (g < cnt_e) ? wgt[e * T_TOKENS + j] : 0.f;   // pad lanes add 0
  }

  // ---- TDM gather: 16 token rows (2KB bf16 each) -> xs, issued by wave 0 ----
  if (w == 0) {
    int tokv[TILE_M];
#pragma unroll
    for (int i = 0; i < TILE_M; ++i) {               // uniform scalar loads
      int g = m * TILE_M + i;
      int j = min(g, cnt_e - 1);
      tokv[i] = idx[e * T_TOKENS + j];
    }
    // D# group 0: count=1, gather_mode=1 (16-bit idx), lds_addr, global_addr, type=2
    unsigned long long ga =
        ((unsigned long long)(uintptr_t)xb) & 0x01FFFFFFFFFFFFFFull;
    v4u g0;
    g0[0] = 1u | (1u << 31);                         // count=1 | gather_mode
    g0[1] = (u32)(uintptr_t)&xs[0][0];               // LDS byte address
    g0[2] = (u32)ga;                                 // global_addr[31:0]
    g0[3] = (u32)((ga >> 32) & 0x01FFFFFFu) | (2u << 30);  // addr[56:32] | type=2
    // D# group 1: data_size=2B, tensor_dim0=1024, tensor_dim1=4096,
    //             tile_dim0=1024, tile_dim1=16 indices, dim0_stride=1024
    v8u g1;
    g1[0] = (1u << 16);                              // data_size = 2 bytes
    g1[1] = (1024u << 16);                           // tensor_dim0[15:0]
    g1[2] = (4096u << 16);                           // tensor_dim0 hi | tensor_dim1 lo
    g1[3] = (1024u << 16);                           // tensor_dim1 hi | tile_dim0
    g1[4] = 16u;                                     // tile_dim1 = #valid indices
    g1[5] = 1024u;                                   // tensor_dim0_stride lo
    g1[6] = 0u;
    g1[7] = 0u;
    // D# groups 2/3: 16 row indices, 16-bit packed
    v4u g2, g3;
#pragma unroll
    for (int i = 0; i < 4; ++i) {
      g2[i] = ((u32)tokv[2 * i]     & 0xFFFFu) | (((u32)tokv[2 * i + 1]) << 16);
      g3[i] = ((u32)tokv[8 + 2 * i] & 0xFFFFu) | (((u32)tokv[9 + 2 * i]) << 16);
    }
    asm volatile("tensor_load_to_lds %0, %1, %2, %3"
                 :: "s"(g0), "s"(g1), "s"(g2), "s"(g3)
                 : "memory");
    __builtin_amdgcn_s_wait_tensorcnt(0);            // TENSORcnt is per-wave
  }
  __syncthreads();                                   // publish xs to all waves

  const float* W1e = W1 + (long)e * D_FF * D_MODEL;
  const float* W3e = W3 + (long)e * D_FF * D_MODEL;
  const float* W2e = W2 + (long)e * D_MODEL * D_FF;

  v8f acc[8];                                        // y[16, w*128 .. +128)
#pragma unroll
  for (int nt = 0; nt < 8; ++nt) acc[nt] = (v8f)0.f;

  for (int fc = 0; fc < D_FF; fc += FCHUNK) {
    // ---- stage 1: h[16, fc + w*32 .. +32) = silu(x W1^T) * (x W3^T) ----
    const int fbase = fc + w * 32;
    v8f c1a = (v8f)0.f, c1b = (v8f)0.f, c3a = (v8f)0.f, c3b = (v8f)0.f;
    for (int kk = 0; kk < D_MODEL; kk += 32) {
      v16bf a   = load_a_lds(&xs[0][0], D_MODEL, kk, lane);
      v16bf b1a = load_b_gf32(W1e + (long)fbase * D_MODEL,        D_MODEL, kk, lane);
      v16bf b1b = load_b_gf32(W1e + (long)(fbase + 16) * D_MODEL, D_MODEL, kk, lane);
      v16bf b3a = load_b_gf32(W3e + (long)fbase * D_MODEL,        D_MODEL, kk, lane);
      v16bf b3b = load_b_gf32(W3e + (long)(fbase + 16) * D_MODEL, D_MODEL, kk, lane);
      __builtin_prefetch(W2e + (long)(w * 128) * D_FF + fc + kk, 0, 1);
      c1a = WMMA_BF16(a, b1a, c1a);
      c1b = WMMA_BF16(a, b1b, c1b);
      c3a = WMMA_BF16(a, b3a, c3a);
      c3b = WMMA_BF16(a, b3b, c3b);
    }
    {
      const int n = lane & 15, half = lane >> 4;
#pragma unroll
      for (int r = 0; r < 8; ++r) {
        int mrow = r + 8 * half;                     // C layout: VGPR r -> M=r / r+8
        hs[mrow][w * 32 + n]      = (__bf16)(fast_silu(c1a[r]) * c3a[r]);
        hs[mrow][w * 32 + 16 + n] = (__bf16)(fast_silu(c1b[r]) * c3b[r]);
      }
    }
    __syncthreads();

    // ---- stage 2: y[16, w*128 .. +128) += h[16,256] * W2^T chunk ----
    for (int k2 = 0; k2 < FCHUNK; k2 += 32) {
      v16bf ah = load_a_lds(&hs[0][0], FCHUNK, k2, lane);
#pragma unroll
      for (int nt = 0; nt < 8; ++nt) {
        int col0 = w * 128 + nt * 16;
        v16bf b = load_b_gf32(W2e + (long)col0 * D_FF + fc, D_FF, k2, lane);
        acc[nt] = WMMA_BF16(ah, b, acc[nt]);
      }
    }
    __syncthreads();                                 // hs reused next chunk
  }

  // ---- combine-weighted scatter (two experts per token -> atomic add) ----
  {
    const int n = lane & 15, half = lane >> 4;
#pragma unroll
    for (int nt = 0; nt < 8; ++nt) {
      int col = w * 128 + nt * 16 + n;
#pragma unroll
      for (int r = 0; r < 8; ++r) {
        int mrow = r + 8 * half;
        float v = wt_s[mrow] * acc[nt][r];
        atomicAdd(&out[(long)tok_s[mrow] * D_MODEL + col], v);
      }
    }
  }
}

// ---------------- kernel 3: aux loss ----------------
__global__ void moe_aux(const float* __restrict__ imp, const int* __restrict__ cnt,
                        float* __restrict__ aux) {
  if (threadIdx.x == 0) {
    float s = 0.f;
#pragma unroll
    for (int e = 0; e < N_EXPERTS; ++e) s += imp[e] * (float)cnt[e];
    float tt = (float)T_TOKENS * (float)T_TOKENS + 1e-6f;
    aux[0] = s * ((float)N_EXPERTS / tt);
  }
}

// ---------------- launcher ----------------
extern "C" void kernel_launch(void* const* d_in, const int* in_sizes, int n_in,
                              void* d_out, int out_size, void* d_ws, size_t ws_size,
                              hipStream_t stream) {
  const float* x  = (const float*)d_in[0];
  const float* Wr = (const float*)d_in[1];
  const float* W1 = (const float*)d_in[2];
  const float* W2 = (const float*)d_in[3];
  const float* W3 = (const float*)d_in[4];
  float* out = (float*)d_out;                       // [T*D] + 1 (aux)

  char* wsb = (char*)d_ws;
  int*    cnt = (int*)wsb;                           // 8 ints
  float*  imp = (float*)(wsb + 64);                  // 8 floats
  int*    idx = (int*)(wsb + 128);                   // 8*T ints
  float*  wgt = (float*)(wsb + 128 + (size_t)N_EXPERTS * T_TOKENS * 4);
  __bf16* xb  = (__bf16*)(wsb + (1u << 19));         // T*D bf16 = 8 MB

  const int n4 = (T_TOKENS * D_MODEL) / 4;
  // 0) zero output + counters ; convert x to bf16 staging buffer
  moe_zero<<<(n4 + 255) / 256, 256, 0, stream>>>(out, cnt, imp, n4);
  moe_cvt_bf16<<<(n4 + 255) / 256, 256, 0, stream>>>(x, xb, n4);
  // 1) router (2 tokens per 64-thread block)
  moe_router<<<T_TOKENS / 2, 64, 0, stream>>>(x, Wr, cnt, idx, wgt, imp);
  // 2) fused expert FFN over compacted token lists
  {
    dim3 grid(N_EXPERTS * (T_TOKENS / TILE_M));
    moe_expert_ffn<<<grid, 256, 0, stream>>>(xb, W1, W2, W3, cnt, idx, wgt, out);
  }
  // 3) aux loss
  moe_aux<<<1, 32, 0, stream>>>(imp, cnt, out + (size_t)T_TOKENS * D_MODEL);
}
